// SpanEmo_18210661335640
// MI455X (gfx1250) — matrix operand order repaired and checked by reading, not compiled, not run on hardware
//
#include <hip/hip_runtime.h>
#include <hip/hip_bf16.h>
#include <math.h>

// Problem constants from the reference
#define B_   256
#define S_   128
#define H_   768
#define C_   11
#define M_   (B_ * C_)      // 2816 = 176 * 16 rows actually needed
#define ALPHA_ 0.2f

typedef __attribute__((ext_vector_type(8)))  _Float16 v8h;
typedef __attribute__((ext_vector_type(16))) _Float16 v16h;
typedef __attribute__((ext_vector_type(8)))  float    v8f;

// ---------------------------------------------------------------------------
// Kernel 1: gather the 11 needed sequence positions per batch, convert to f16.
// A[m][k] = (f16) hidden[b, label_idxs[c], k],  m = b*11 + c
// ---------------------------------------------------------------------------
__global__ __launch_bounds__(256) void gather_convert_A(
    const float* __restrict__ hidden,
    const long long* __restrict__ label_idxs,
    _Float16* __restrict__ Ah)
{
    const int m = blockIdx.x;           // 0 .. 2815
    const int b = m / C_;
    const int c = m - b * C_;
    const int s = (int)label_idxs[c];
    const float* __restrict__ src = hidden + ((size_t)b * S_ + s) * H_;
    _Float16* __restrict__ dst = Ah + (size_t)m * H_;
    for (int k = threadIdx.x; k < H_; k += 256)
        dst[k] = (_Float16)src[k];
}

// ---------------------------------------------------------------------------
// Kernel 2: W1 (K-major, [k][n]) -> Bt (N-major f16, [n][k]); also zero logits.
// ---------------------------------------------------------------------------
__global__ __launch_bounds__(256) void transpose_convert_W(
    const float* __restrict__ W1,
    _Float16* __restrict__ Bt,
    float* __restrict__ logits)
{
    const int n = blockIdx.x;           // 0 .. 767
    for (int k = threadIdx.x; k < H_; k += 256)
        Bt[(size_t)n * H_ + k] = (_Float16)W1[(size_t)k * H_ + n];
    if (n < C_)                         // 11 blocks * 256 threads == 2816
        logits[n * 256 + threadIdx.x] = 0.0f;
}

// ---------------------------------------------------------------------------
// Kernel 3: WMMA GEMM (2816 x 768 x 768, f16 in / f32 accum) fused with
// +b1 -> tanh -> dot(W2) row-reduction.  Block = 128 threads = 4 wave32.
// Block tile: 16 (M) x 256 (N); each wave: 16 x 64 via 4 accumulators.
// ---------------------------------------------------------------------------
__global__ __launch_bounds__(128) void gemm_tanh_dot(
    const _Float16* __restrict__ Ah,    // [M_][H_]
    const _Float16* __restrict__ Bt,    // [H_][H_]  N-major (row n = column of W1)
    const float* __restrict__ b1,
    const float* __restrict__ W2,
    float* __restrict__ logits)         // [M_] fp32, atomically accumulated
{
    const int lane = threadIdx.x & 31;
    const int wave = threadIdx.x >> 5;              // 0..3
    const int m0   = blockIdx.x * 16;               // M tile base
    const int n0   = blockIdx.y * 256 + wave * 64;  // this wave's N base

    const int half = lane >> 4;                     // 0 or 1 (lane group)
    const int l    = lane & 15;

    // A fragment (16x32 f16, ISA layout):
    //   lanes 0-15:  row=l, v0..3 = K 0..7,  v4..7 = K 16..23
    //   lanes 16-31: row=l, v0..3 = K 8..15, v4..7 = K 24..31
    const _Float16* __restrict__ arow = Ah + (size_t)(m0 + l) * H_ + half * 8;

    v8f acc[4] = {}, z = {};
    acc[0] = z; acc[1] = z; acc[2] = z; acc[3] = z;

    for (int k = 0; k < H_; k += 32) {
        __builtin_prefetch(arow + k + 64, 0, 0);    // global_prefetch_b8
        v8h a_lo = *(const v8h*)(arow + k);
        v8h a_hi = *(const v8h*)(arow + k + 16);
        v16h afrag = __builtin_shufflevector(a_lo, a_hi,
            0,1,2,3,4,5,6,7,8,9,10,11,12,13,14,15);
#pragma unroll
        for (int t = 0; t < 4; ++t) {
            // B fragment (32x16 f16): lanes 0-15 hold col n, K 0..7, 16..23;
            // lanes 16-31 hold col n, K 8..15, 24..31 (same order as A).
            const _Float16* __restrict__ bcol =
                Bt + (size_t)(n0 + t * 16 + l) * H_ + half * 8;
            v8h b_lo = *(const v8h*)(bcol + k);
            v8h b_hi = *(const v8h*)(bcol + k + 16);
            v16h bfrag = __builtin_shufflevector(b_lo, b_hi,
                0,1,2,3,4,5,6,7,8,9,10,11,12,13,14,15);
            acc[t] = __builtin_amdgcn_wmma_f32_16x16x32_f16(
                /*neg_a=*/false, afrag, /*neg_b=*/false, bfrag,
                /*c_mod=*/(short)0, acc[t],
                /*reuse_a=*/false, /*reuse_b=*/false);
        }
    }

    // Epilogue. C/D layout: vgpr j, lanes 0-15 -> (row=j,   col=l)
    //                                lanes16-31 -> (row=j+8, col=l)
    // Fuse: h = tanh(acc + b1[n]); partial = h * W2[n]; reduce cols; atomicAdd.
#pragma unroll
    for (int j = 0; j < 8; ++j) {
        float s = 0.0f;
#pragma unroll
        for (int t = 0; t < 4; ++t) {
            const int n = n0 + t * 16 + l;
            const float hv = tanhf(acc[t][j] + b1[n]);
            s += hv * W2[n];
        }
        // reduce the 16 columns within each 16-lane group (wave32)
        s += __shfl_xor(s, 8, 16);
        s += __shfl_xor(s, 4, 16);
        s += __shfl_xor(s, 2, 16);
        s += __shfl_xor(s, 1, 16);
        if (l == 0)
            atomicAdd(&logits[m0 + j + 8 * half], s);
    }
}

// ---------------------------------------------------------------------------
// Kernel 4: loss. One block of 256 threads; thread b handles batch row b.
// ---------------------------------------------------------------------------
__global__ __launch_bounds__(256) void loss_kernel(
    const float* __restrict__ logits,   // [B_][C_] (pre-b2)
    const float* __restrict__ targets,  // [B_][C_]
    const float* __restrict__ b2p,
    float* __restrict__ out)
{
    __shared__ float s_cel[256];
    __shared__ float s_cl[256];
    const int b  = threadIdx.x;
    const float b2 = b2p[0];

    float cel = 0.0f, s_z = 0.0f, s_o = 0.0f;
    int   n_z = 0,    n_o = 0;
#pragma unroll
    for (int c = 0; c < C_; ++c) {
        const float lv = logits[b * C_ + c] + b2;
        const float t  = targets[b * C_ + c];
        cel += fmaxf(lv, 0.0f) - lv * t + log1pf(expf(-fabsf(lv)));
        const float p = 1.0f / (1.0f + expf(-lv));
        if (t > 0.0f) { s_o += expf(-p); ++n_o; }
        else          { s_z += expf(p);  ++n_z; }
    }
    const float denom  = fmaxf((float)(n_z * n_o), 1.0f);
    const float cl_row = (n_o > 0) ? (s_z * s_o / denom) : 0.0f;

    s_cel[b] = cel;
    s_cl[b]  = cl_row;
    __syncthreads();
    for (int stride = 128; stride >= 1; stride >>= 1) {
        if (b < stride) {
            s_cel[b] += s_cel[b + stride];
            s_cl[b]  += s_cl[b + stride];
        }
        __syncthreads();
    }
    if (b == 0)
        out[0] = (1.0f - ALPHA_) * (s_cel[0] / (float)(B_ * C_))
               + ALPHA_          * (s_cl[0]  / (float)B_);
}

// ---------------------------------------------------------------------------
extern "C" void kernel_launch(void* const* d_in, const int* in_sizes, int n_in,
                              void* d_out, int out_size, void* d_ws, size_t ws_size,
                              hipStream_t stream)
{
    const float*     hidden  = (const float*)d_in[0];
    const float*     W1      = (const float*)d_in[1];
    const float*     b1      = (const float*)d_in[2];
    const float*     W2      = (const float*)d_in[3];
    const float*     b2      = (const float*)d_in[4];
    const float*     targets = (const float*)d_in[5];
    const long long* lidx    = (const long long*)d_in[6];

    char* ws = (char*)d_ws;
    _Float16* Ah     = (_Float16*)ws;                               // 2816*768*2 B
    _Float16* Bt     = (_Float16*)(ws + (size_t)M_ * H_ * 2);       // 768*768*2 B
    float*    logits = (float*)   (ws + (size_t)M_ * H_ * 2
                                      + (size_t)H_ * H_ * 2);       // 2816*4 B
    float*    out    = (float*)d_out;

    gather_convert_A   <<<M_,  256, 0, stream>>>(hidden, lidx, Ah);
    transpose_convert_W<<<H_,  256, 0, stream>>>(W1, Bt, logits);
    gemm_tanh_dot      <<<dim3(M_ / 16, H_ / 256), 128, 0, stream>>>(Ah, Bt, b1, W2, logits);
    loss_kernel        <<<1,   256, 0, stream>>>(logits, targets, b2, out);
}
